// FrequencySparseAttention_18459769438348
// MI455X (gfx1250) — matrix operand order repaired and checked by reading, not compiled, not run on hardware
//
#include <hip/hip_runtime.h>

#define B_  4
#define L_  2048
#define D_  1024
#define H_  16
#define DH  64
#define TAU 0.1f

typedef __attribute__((ext_vector_type(16))) __bf16        v16bf;
typedef __attribute__((ext_vector_type(8)))  __bf16        v8bf;
typedef __attribute__((ext_vector_type(8)))  float         v8f;
typedef __attribute__((ext_vector_type(8)))  unsigned int  u32x8;
typedef __attribute__((ext_vector_type(4)))  unsigned int  u32x4;
typedef __attribute__((ext_vector_type(4)))  int           v4i;

// ---------------- helpers ----------------

__device__ __forceinline__ unsigned short f2bf(float f) {
    return __builtin_bit_cast(unsigned short, (__bf16)f);
}

__device__ __forceinline__ v8f zero8() {
    v8f z = {0.f, 0.f, 0.f, 0.f, 0.f, 0.f, 0.f, 0.f};
    return z;
}

__device__ __forceinline__ v8f wmma_bf16(v16bf a, v16bf b, v8f c) {
    // D = A(16x32 bf16) * B(32x16 bf16) + C(16x16 f32)
    return __builtin_amdgcn_wmma_f32_16x16x32_bf16(
        false, a, false, b, (short)0, c, false, false);
}

// A-fragment (16x32, 16-bit). Lane-group0 holds K=[0..15]; group1
// K=[16..31]. Caller pre-adds the per-lane-group offset ((lane>>4)*16).
__device__ __forceinline__ v16bf load_a16(const unsigned short* rowp) {
    const unsigned int* p = (const unsigned int*)rowp;
    u32x8 t;
#pragma unroll
    for (int j = 0; j < 4; ++j) t[j] = p[j];
#pragma unroll
    for (int j = 0; j < 4; ++j) t[4 + j] = p[4 + j];
    return __builtin_bit_cast(v16bf, t);
}

// B-fragment (32x16, 16-bit): lane%16 = N; caller pre-adds per-lane K offset
// ((lane>>4)*16); 16 values contiguous at rowp[0..15].
__device__ __forceinline__ v16bf load_b16(const unsigned short* rowp) {
    const unsigned int* p = (const unsigned int*)rowp;
    u32x8 t;
#pragma unroll
    for (int j = 0; j < 8; ++j) t[j] = p[j];
    return __builtin_bit_cast(v16bf, t);
}

// ---------------- kernel 0: f32 -> bf16 bulk convert ----------------
__global__ void __launch_bounds__(256) cvt_kernel(const float* __restrict__ in,
                                                  unsigned short* __restrict__ out, int n8) {
    const int i = blockIdx.x * 256 + threadIdx.x;
    if (i >= n8) return;
    const float4* p = (const float4*)(in) + 2 * (size_t)i;
    float4 a = p[0], b = p[1];
    v8bf r;
    r[0] = (__bf16)a.x; r[1] = (__bf16)a.y; r[2] = (__bf16)a.z; r[3] = (__bf16)a.w;
    r[4] = (__bf16)b.x; r[5] = (__bf16)b.y; r[6] = (__bf16)b.z; r[7] = (__bf16)b.w;
    ((u32x4*)out)[i] = __builtin_bit_cast(u32x4, r);
}

// ---------------- kernel 1: energy gate ----------------
__global__ void __launch_bounds__(256) gate_kernel(const float* __restrict__ e,
                                                   float* __restrict__ gate) {
    const int wave = threadIdx.x >> 5;
    const int lane = threadIdx.x & 31;
    const int row  = blockIdx.x * 8 + wave;
    const float4* p = (const float4*)(e + (size_t)row * D_);
    float s = 0.f;
#pragma unroll
    for (int i = 0; i < D_ / 128; ++i) {
        float4 x = p[lane + i * 32];
        s += fabsf(x.x) + fabsf(x.y) + fabsf(x.z) + fabsf(x.w);
    }
#pragma unroll
    for (int m = 16; m >= 1; m >>= 1) s += __shfl_xor(s, m, 32);
    if (lane == 0) {
        float mean = s * (1.0f / D_);
        gate[row] = 1.0f / (1.0f + __expf(-(mean - TAU) * 10.0f));
    }
}

// ---------------- GEMM core: wave tile 32(M) x 64(N), ping-pong pipelined ----------------
// Block = 8 waves -> block tile 256 x 64. All operands bf16.
struct GemmAcc { v8f a[2][4]; };

__device__ __forceinline__ void gemm_core(const unsigned short* __restrict__ X,
                                          const unsigned short* __restrict__ W,
                                          int m0, int n0, int lane, GemmAcc& g) {
    const int aoff = (lane >> 4) << 4;
    const int boff = (lane >> 4) << 4;
    const unsigned short* ar0 = X + (size_t)(m0 + (lane & 15)) * D_ + aoff;
    const unsigned short* ar1 = ar0 + (size_t)16 * D_;
    const unsigned short* wr  = W + (size_t)(n0 + (lane & 15)) * D_ + boff;

#pragma unroll
    for (int mi = 0; mi < 2; ++mi)
#pragma unroll
        for (int nf = 0; nf < 4; ++nf) g.a[mi][nf] = zero8();

    // ping-pong fragment buffers (no copies between stages)
    v16bf a0_0 = load_a16(ar0), a1_0 = load_a16(ar1);
    v16bf b_0[4];
#pragma unroll
    for (int nf = 0; nf < 4; ++nf) b_0[nf] = load_b16(wr + (size_t)nf * 16 * D_);
    v16bf a0_1, a1_1, b_1[4];

#pragma unroll 1
    for (int k0 = 0; k0 < D_ - 64; k0 += 64) {
        // stage 1 loads (k0+32) ahead of stage-0 compute
        a0_1 = load_a16(ar0 + k0 + 32);
        a1_1 = load_a16(ar1 + k0 + 32);
#pragma unroll
        for (int nf = 0; nf < 4; ++nf) b_1[nf] = load_b16(wr + (size_t)nf * 16 * D_ + k0 + 32);
#pragma unroll
        for (int nf = 0; nf < 4; ++nf) {
            g.a[0][nf] = wmma_bf16(a0_0, b_0[nf], g.a[0][nf]);
            g.a[1][nf] = wmma_bf16(a1_0, b_0[nf], g.a[1][nf]);
        }
        // stage 0 loads (k0+64) ahead of stage-1 compute
        a0_0 = load_a16(ar0 + k0 + 64);
        a1_0 = load_a16(ar1 + k0 + 64);
#pragma unroll
        for (int nf = 0; nf < 4; ++nf) b_0[nf] = load_b16(wr + (size_t)nf * 16 * D_ + k0 + 64);
#pragma unroll
        for (int nf = 0; nf < 4; ++nf) {
            g.a[0][nf] = wmma_bf16(a0_1, b_1[nf], g.a[0][nf]);
            g.a[1][nf] = wmma_bf16(a1_1, b_1[nf], g.a[1][nf]);
        }
    }
    // tail: k = D_-64 in buf0, load and do k = D_-32 in buf1
    a0_1 = load_a16(ar0 + D_ - 32);
    a1_1 = load_a16(ar1 + D_ - 32);
#pragma unroll
    for (int nf = 0; nf < 4; ++nf) b_1[nf] = load_b16(wr + (size_t)nf * 16 * D_ + D_ - 32);
#pragma unroll
    for (int nf = 0; nf < 4; ++nf) {
        g.a[0][nf] = wmma_bf16(a0_0, b_0[nf], g.a[0][nf]);
        g.a[1][nf] = wmma_bf16(a1_0, b_0[nf], g.a[1][nf]);
    }
#pragma unroll
    for (int nf = 0; nf < 4; ++nf) {
        g.a[0][nf] = wmma_bf16(a0_1, b_1[nf], g.a[0][nf]);
        g.a[1][nf] = wmma_bf16(a1_1, b_1[nf], g.a[1][nf]);
    }
}

// ---------------- kernel 2: input projection -> bf16 (B,H,L,DH) ----------------
__global__ void __launch_bounds__(256, 1) proj_kernel(const unsigned short* __restrict__ X,
                                                      const unsigned short* __restrict__ W,
                                                      const float* __restrict__ bias,
                                                      unsigned short* __restrict__ out) {
    const int wave = threadIdx.x >> 5;
    const int lane = threadIdx.x & 31;
    const int m0   = blockIdx.x * 256 + wave * 32;
    const int n0   = blockIdx.y * 64;

    GemmAcc g;
    gemm_core(X, W, m0, n0, lane, g);

    // C/D layout: VGPR v -> row base + v + 8*(lane>=16); lane%16 -> N.
#pragma unroll
    for (int nf = 0; nf < 4; ++nf) {
        const int n  = n0 + nf * 16 + (lane & 15);
        const float bv = bias[n];
        const int h  = n >> 6, dh = n & 63;
#pragma unroll
        for (int mi = 0; mi < 2; ++mi) {
            const int mrow = m0 + mi * 16 + ((lane >> 4) << 3);
#pragma unroll
            for (int v = 0; v < 8; ++v) {
                const int m = mrow + v;
                const int b = m >> 11, l = m & (L_ - 1);
                out[(((size_t)(b * H_ + h)) * L_ + l) * DH + dh] = f2bf(g.a[mi][nf][v] + bv);
            }
        }
    }
}

// ---------------- kernel 4: output projection -> f32 (B,L,D) ----------------
__global__ void __launch_bounds__(256, 1) oproj_kernel(const unsigned short* __restrict__ A,
                                                       const unsigned short* __restrict__ W,
                                                       const float* __restrict__ bias,
                                                       float* __restrict__ out) {
    const int wave = threadIdx.x >> 5;
    const int lane = threadIdx.x & 31;
    const int m0   = blockIdx.x * 256 + wave * 32;
    const int n0   = blockIdx.y * 64;

    GemmAcc g;
    gemm_core(A, W, m0, n0, lane, g);

#pragma unroll
    for (int nf = 0; nf < 4; ++nf) {
        const int n  = n0 + nf * 16 + (lane & 15);
        const float bv = bias[n];
#pragma unroll
        for (int mi = 0; mi < 2; ++mi) {
            const int mrow = m0 + mi * 16 + ((lane >> 4) << 3);
#pragma unroll
            for (int v = 0; v < 8; ++v)
                out[(size_t)(mrow + v) * D_ + n] = g.a[mi][nf][v] + bv;
        }
    }
}

// ---------------- kernel 3: gated flash attention ----------------
__global__ void __launch_bounds__(256, 1) attn_kernel(const unsigned short* __restrict__ Qb,
                                                      const unsigned short* __restrict__ Kb,
                                                      const unsigned short* __restrict__ Vb,
                                                      const float* __restrict__ gate,
                                                      unsigned short* __restrict__ Ob) {
    const int bh   = blockIdx.x;
    const int b    = bh >> 4;
    const int h    = bh & 15;
    const int wave = threadIdx.x >> 5;
    const int lane = threadIdx.x & 31;
    const int q0   = blockIdx.y * 128 + wave * 16;

    const unsigned short* Qp = Qb + (size_t)bh * L_ * DH;
    const unsigned short* Kp = Kb + (size_t)bh * L_ * DH;
    const unsigned short* Vp = Vb + (size_t)bh * L_ * DH;
    const float* gp = gate + (size_t)b * L_;

    __shared__ __align__(16) unsigned short sK[32 * DH];       // K tile, row=key
    __shared__ __align__(16) unsigned short sVt[DH * 32];      // V tile transposed
    __shared__ __align__(16) unsigned short sP[8][16 * 32];    // per-wave P staging

    const int aoff = (lane >> 4) << 4;
    const int boff = (lane >> 4) << 4;

    v16bf qa[2];
#pragma unroll
    for (int kk = 0; kk < 2; ++kk)
        qa[kk] = load_a16(Qp + (size_t)(q0 + (lane & 15)) * DH + kk * 32 + aoff);

    v8f oacc[4] = {zero8(), zero8(), zero8(), zero8()};
    float mrow[8], lrow[8];
#pragma unroll
    for (int v = 0; v < 8; ++v) { mrow[v] = -1e30f; lrow[v] = 0.f; }

    for (int key0 = 0; key0 < L_; key0 += 32) {
        __syncthreads();

        // --- stage K tile (async DMA to LDS), V tile transposed ---
#if defined(__gfx1250__) && __has_builtin(__builtin_amdgcn_global_load_async_to_lds_b128)
        {
            typedef __attribute__((address_space(1))) v4i* gp4_t;   // global int4*
            typedef __attribute__((address_space(3))) v4i* lp4_t;   // LDS int4*
            const unsigned short* gk = Kp + (size_t)key0 * DH + threadIdx.x * 8;  // 16B/lane
            const unsigned lds = (unsigned)(uintptr_t)sK + threadIdx.x * 16u;
            __builtin_amdgcn_global_load_async_to_lds_b128(
                (gp4_t)(uintptr_t)gk, (lp4_t)(uintptr_t)lds, 0, 0);
        }
#else
        {
            const unsigned int* gKu = (const unsigned int*)(Kp + (size_t)key0 * DH);
            unsigned int* sKu = (unsigned int*)sK;
            for (int i = threadIdx.x; i < 32 * DH / 2; i += 256) sKu[i] = gKu[i];
        }
#endif
        {
            const unsigned int* gVu = (const unsigned int*)(Vp + (size_t)key0 * DH);
            for (int i = threadIdx.x; i < 32 * DH / 2; i += 256) {
                const unsigned int vw = gVu[i];
                const int key = (2 * i) / DH, d = (2 * i) % DH;
                sVt[d * 32 + key]       = (unsigned short)(vw & 0xFFFFu);
                sVt[(d + 1) * 32 + key] = (unsigned short)(vw >> 16);
            }
        }
        if (key0 + 32 < L_) {   // prefetch next tiles into cache
            __builtin_prefetch(Kp + (size_t)(key0 + 32) * DH + threadIdx.x * 8, 0, 1);
            __builtin_prefetch(Vp + (size_t)(key0 + 32) * DH + threadIdx.x * 8, 0, 1);
        }
#if defined(__gfx1250__) && __has_builtin(__builtin_amdgcn_global_load_async_to_lds_b128)
        asm volatile("s_wait_asynccnt 0" ::: "memory");
#endif
        __syncthreads();

        // --- scores: two 16x16 tiles ---
        v8f s0 = zero8(), s1 = zero8();
#pragma unroll
        for (int kk = 0; kk < 2; ++kk) {
            s0 = wmma_bf16(qa[kk], load_b16(sK + (0  + (lane & 15)) * DH + kk * 32 + boff), s0);
            s1 = wmma_bf16(qa[kk], load_b16(sK + (16 + (lane & 15)) * DH + kk * 32 + boff), s1);
        }

        const float g0 = gp[key0 + (lane & 15)];
        const float g1 = gp[key0 + 16 + (lane & 15)];
        unsigned short* sPw = sP[wave];
        const int prow = ((lane >> 4) << 3);

        // --- online softmax (gate folded into P, ungated denominator) ---
#pragma unroll
        for (int v = 0; v < 8; ++v) {
            float x0 = s0[v] * 0.125f;
            float x1 = s1[v] * 0.125f;
            float mx = fmaxf(x0, x1);
#pragma unroll
            for (int d = 1; d < 16; d <<= 1) mx = fmaxf(mx, __shfl_xor(mx, d, 16));
            const float mnew  = fmaxf(mrow[v], mx);
            const float alpha = __expf(mrow[v] - mnew);
            const float p0 = __expf(x0 - mnew);
            const float p1 = __expf(x1 - mnew);
            float rs = p0 + p1;
#pragma unroll
            for (int d = 1; d < 16; d <<= 1) rs += __shfl_xor(rs, d, 16);
            lrow[v] = lrow[v] * alpha + rs;
            mrow[v] = mnew;
#pragma unroll
            for (int nf = 0; nf < 4; ++nf) oacc[nf][v] *= alpha;
            const int r = (prow + v) * 32 + (lane & 15);
            sPw[r]      = f2bf(p0 * g0);
            sPw[r + 16] = f2bf(p1 * g1);
        }
        asm volatile("s_wait_dscnt 0" ::: "memory");   // LDS RAW within wave

        // --- P(16x32) @ V(32x64) ---
        v16bf pa = load_a16(sPw + (lane & 15) * 32 + aoff);
#pragma unroll
        for (int nf = 0; nf < 4; ++nf)
            oacc[nf] = wmma_bf16(pa, load_b16(sVt + (nf * 16 + (lane & 15)) * 32 + boff), oacc[nf]);
    }

    // normalize, store bf16 (B,L,D)
#pragma unroll
    for (int nf = 0; nf < 4; ++nf) {
        const int col = h * DH + nf * 16 + (lane & 15);
#pragma unroll
        for (int v = 0; v < 8; ++v) {
            const int qr = q0 + v + ((lane >> 4) << 3);
            Ob[((size_t)b * L_ + qr) * D_ + col] = f2bf(oacc[nf][v] / lrow[v]);
        }
    }
}

// ---------------- launch ----------------
extern "C" void kernel_launch(void* const* d_in, const int* in_sizes, int n_in,
                              void* d_out, int out_size, void* d_ws, size_t ws_size,
                              hipStream_t stream) {
    const float* q_x = (const float*)d_in[0];
    const float* k_x = (const float*)d_in[1];
    const float* v_x = (const float*)d_in[2];
    const float* ec  = (const float*)d_in[3];
    const float* Wq  = (const float*)d_in[4];
    const float* bq  = (const float*)d_in[5];
    const float* Wk  = (const float*)d_in[6];
    const float* bk  = (const float*)d_in[7];
    const float* Wv  = (const float*)d_in[8];
    const float* bv  = (const float*)d_in[9];
    const float* Wo  = (const float*)d_in[10];
    const float* bo  = (const float*)d_in[11];

    const size_t nTok  = (size_t)B_ * L_;          // 8192
    const size_t nElem = nTok * D_;                // 8.39M activations
    const size_t nW    = (size_t)D_ * D_;          // 1.05M weights

    char* ws = (char*)d_ws;
    float*          gate = (float*)ws;                       // 32KB (64KB slot)
    unsigned short* Wqb  = (unsigned short*)(ws + (64 << 10));
    unsigned short* Wkb  = Wqb + nW;
    unsigned short* Wvb  = Wkb + nW;
    unsigned short* Wob  = Wvb + nW;
    unsigned short* Xb   = Wob + nW;     // staging; reused as attention output
    unsigned short* Qb   = Xb + nElem;
    unsigned short* Kb   = Qb + nElem;
    unsigned short* Vb   = Kb + nElem;
    unsigned short* Ab   = Xb;

    const int gCvtW = (int)(nW / 8 / 256);
    const int gCvtX = (int)(nElem / 8 / 256);
    dim3 gGemm(nTok / 256, D_ / 64);               // (32, 16)

    cvt_kernel<<<gCvtW, 256, 0, stream>>>(Wq, Wqb, (int)(nW / 8));
    cvt_kernel<<<gCvtW, 256, 0, stream>>>(Wk, Wkb, (int)(nW / 8));
    cvt_kernel<<<gCvtW, 256, 0, stream>>>(Wv, Wvb, (int)(nW / 8));
    cvt_kernel<<<gCvtW, 256, 0, stream>>>(Wo, Wob, (int)(nW / 8));

    gate_kernel<<<nTok / 8, 256, 0, stream>>>(ec, gate);

    cvt_kernel<<<gCvtX, 256, 0, stream>>>(q_x, Xb, (int)(nElem / 8));
    proj_kernel<<<gGemm, 256, 0, stream>>>(Xb, Wqb, bq, Qb);
    cvt_kernel<<<gCvtX, 256, 0, stream>>>(k_x, Xb, (int)(nElem / 8));
    proj_kernel<<<gGemm, 256, 0, stream>>>(Xb, Wkb, bk, Kb);
    cvt_kernel<<<gCvtX, 256, 0, stream>>>(v_x, Xb, (int)(nElem / 8));
    proj_kernel<<<gGemm, 256, 0, stream>>>(Xb, Wvb, bv, Vb);

    attn_kernel<<<dim3(B_ * H_, L_ / 128), 256, 0, stream>>>(Qb, Kb, Vb, gate, Ab);

    oproj_kernel<<<gGemm, 256, 0, stream>>>(Ab, Wob, bo, (float*)d_out);
}
